// EModel_80719615361428
// MI455X (gfx1250) — matrix operand order repaired and checked by reading, not compiled, not run on hardware
//
#include <hip/hip_runtime.h>
#include <cmath>

#define NN 100000
#define EE 1600000
#define CC 128
#define HH 128
#define GG 256
#define NCLS 10
#define FDIM (6 * HH) /* 768 */
#define MAXD 128
#define NEG_SLOPE 0.2f

typedef __bf16 v16bf __attribute__((ext_vector_type(16)));
typedef __bf16 v8bf  __attribute__((ext_vector_type(8)));
typedef float  v8f   __attribute__((ext_vector_type(8)));
typedef unsigned short u16;

__device__ __forceinline__ u16 f2bf(float f) {
  unsigned u = __float_as_uint(f);
  unsigned r = 0x7FFFu + ((u >> 16) & 1u);
  return (u16)((u + r) >> 16);
}

// ---------------------------------------------------------------- utilities
__global__ void k_zero_f(float* p, long long n) {
  for (long long i = blockIdx.x * (long long)blockDim.x + threadIdx.x; i < n;
       i += (long long)gridDim.x * blockDim.x)
    p[i] = 0.f;
}
__global__ void k_zero_i(int* p, long long n) {
  for (long long i = blockIdx.x * (long long)blockDim.x + threadIdx.x; i < n;
       i += (long long)gridDim.x * blockDim.x)
    p[i] = 0;
}
__global__ void k_f32_to_bf16(const float* __restrict__ in, u16* __restrict__ out,
                              long long n) {
  for (long long i = blockIdx.x * (long long)blockDim.x + threadIdx.x; i < n;
       i += (long long)gridDim.x * blockDim.x)
    out[i] = f2bf(in[i]);
}
// W [K,Ncols] row-major (f32)  ->  Bt [Ncols,K] row-major (bf16)
__global__ void k_transpose_bf16(const float* __restrict__ W, u16* __restrict__ Bt,
                                 int K, int Ncols) {
  long long n = (long long)K * Ncols;
  for (long long i = blockIdx.x * (long long)blockDim.x + threadIdx.x; i < n;
       i += (long long)gridDim.x * blockDim.x) {
    int k = (int)(i / Ncols), c = (int)(i % Ncols);
    Bt[(size_t)c * K + k] = f2bf(W[i]);
  }
}

// ------------------------------------------------------ WMMA bf16 GEMM core
// Cm[M,Ncols] = A[M,K](bf16) * Bt[Ncols,K](bf16, pre-transposed B).
// One wave per 16x16 output tile; K multiple of 32; M,Ncols multiples of 16.
__global__ void k_wmma_gemm(const u16* __restrict__ A, const u16* __restrict__ Bt,
                            float* __restrict__ Cm, int M, int Ncols, int K) {
  const int lane = threadIdx.x & 31;
  const int mrow = lane & 15;
  const int half = lane >> 4;
  const int tm = blockIdx.x, tn = blockIdx.y;
  int m = tm * 16 + mrow;
  if (m >= M) m = M - 1;
  const int n = tn * 16 + mrow;
  const u16* Ar = A + (size_t)m * K;
  const u16* Br = Bt + (size_t)n * K;
  v8f acc = {};
  for (int kc = 0; kc < K; kc += 32) {
    // A 16x32 bf16 layout: lanes 0-15 = M rows with K 0-7 & 16-23,
    // lanes 16-31 same M with K 8-15 & 24-31 (two 16B contiguous loads).
    v8bf a0 = *(const v8bf*)(Ar + kc + half * 8);
    v8bf a1 = *(const v8bf*)(Ar + kc + 16 + half * 8);
    v16bf a, b;
#pragma unroll
    for (int i = 0; i < 8; ++i) { a[i] = a0[i]; a[8 + i] = a1[i]; }
    // B 32x16 bf16 layout: lane = column N, same K split as A
    // (K 0-7 & 16-23 / 8-15 & 24-31, two 16B loads from the transposed weight row).
    v8bf b0 = *(const v8bf*)(Br + kc + half * 8);
    v8bf b1 = *(const v8bf*)(Br + kc + 16 + half * 8);
#pragma unroll
    for (int i = 0; i < 8; ++i) { b[i] = b0[i]; b[8 + i] = b1[i]; }
    acc = __builtin_amdgcn_wmma_f32_16x16x32_bf16(false, a, false, b, (short)0,
                                                  acc, false, false);
  }
  // D layout: lane = N col, VGPR v = M row (v or 8+v per lane-half)
#pragma unroll
  for (int v = 0; v < 8; ++v) {
    int mm = tm * 16 + half * 8 + v;
    if (mm < M) Cm[(size_t)mm * Ncols + (tn * 16 + mrow)] = acc[v];
  }
}

// ------------------------------------------------------------- attention
// s1[n] = x[n,:] . att[0:C],  s2[n] = x[n,:] . att[C:2C]; one wave per node.
__global__ void k_scores(const float* __restrict__ x, const float* __restrict__ att,
                         float* __restrict__ s1, float* __restrict__ s2) {
  int node = blockIdx.x * (blockDim.x >> 5) + (threadIdx.x >> 5);
  int lane = threadIdx.x & 31;
  if (node >= NN) return;
  float a = 0.f, b = 0.f;
  const float* xr = x + (size_t)node * CC;
  for (int c = lane; c < CC; c += 32) {
    float xv = xr[c];
    a += xv * att[c];
    b += xv * att[CC + c];
  }
  for (int off = 16; off; off >>= 1) {
    a += __shfl_down(a, off, 32);
    b += __shfl_down(b, off, 32);
  }
  if (lane == 0) { s1[node] = a; s2[node] = b; }
}

__global__ void k_edge_w(const int* __restrict__ row, const int* __restrict__ col,
                         const float* __restrict__ s1, const float* __restrict__ s2,
                         float* __restrict__ w) {
  int e = blockIdx.x * blockDim.x + threadIdx.x;
  if (e >= EE) return;
  float v = s1[row[e]] + s2[col[e]];
  w[e] = (v >= 0.f) ? v : NEG_SLOPE * v;
}

// ------------------------------------------------------------- CSR by row
__global__ void k_row_count(const int* __restrict__ row, int* __restrict__ rc) {
  int e = blockIdx.x * blockDim.x + threadIdx.x;
  if (e < EE) atomicAdd(&rc[row[e]], 1);
}
__global__ void k_scan_excl(const int* __restrict__ cnts, int* __restrict__ starts,
                            int n) {
  __shared__ int buf[1024];
  __shared__ int carry;
  if (threadIdx.x == 0) carry = 0;
  __syncthreads();
  for (int base = 0; base < n; base += 1024) {
    int i = base + threadIdx.x;
    int v = (i < n) ? cnts[i] : 0;
    buf[threadIdx.x] = v;
    __syncthreads();
    for (int off = 1; off < 1024; off <<= 1) {
      int t = (threadIdx.x >= (unsigned)off) ? buf[threadIdx.x - off] : 0;
      __syncthreads();
      buf[threadIdx.x] += t;
      __syncthreads();
    }
    if (i < n) starts[i] = carry + buf[threadIdx.x] - v;
    int total = buf[1023];
    __syncthreads();
    if (threadIdx.x == 0) carry += total;
    __syncthreads();
  }
}
__global__ void k_csr_fill(const int* __restrict__ row, const int* __restrict__ rstart,
                           int* __restrict__ rcur, int* __restrict__ csr) {
  int e = blockIdx.x * blockDim.x + threadIdx.x;
  if (e >= EE) return;
  int r = row[e];
  int pos = atomicAdd(&rcur[r], 1);
  csr[rstart[r] + pos] = e;
}

// ------------------------------------------------- exact segment sparsemax
// One wave per row: rank-sort desc in LDS, cumsum, support size k, tau.
__global__ void k_sparsemax(const float* __restrict__ wsc, const int* __restrict__ rstart,
                            const int* __restrict__ rc, const int* __restrict__ csr,
                            float* __restrict__ pe) {
  __shared__ float zb[8][MAXD];
  __shared__ float sb[8][MAXD];
  __shared__ float cb[8][MAXD];
  __shared__ float taub[8];
  int wv = threadIdx.x >> 5;
  int lane = threadIdx.x & 31;
  int r = blockIdx.x * 8 + wv;
  int d = 0, st = 0;
  if (r < NN) { d = rc[r]; st = rstart[r]; if (d > MAXD) d = MAXD; }
  for (int j = lane; j < d; j += 32) zb[wv][j] = wsc[csr[st + j]];
  __syncthreads();
  for (int j = lane; j < d; j += 32) {  // stable rank (desc)
    float z = zb[wv][j];
    int rank = 0;
    for (int q = 0; q < d; ++q) {
      float zq = zb[wv][q];
      rank += (zq > z) || (zq == z && q < j);
    }
    sb[wv][rank] = z;
  }
  __syncthreads();
  for (int p = lane; p < d; p += 32) {  // inclusive cumsum of sorted values
    float s = 0.f;
    for (int q = 0; q <= p; ++q) s += sb[wv][q];
    cb[wv][p] = s;
  }
  __syncthreads();
  if (lane == 0) {
    int k = 0;
    float s = 0.f;
    for (int p = 0; p < d; ++p)
      if (1.f + (float)(p + 1) * sb[wv][p] > cb[wv][p]) { ++k; s += sb[wv][p]; }
    taub[wv] = (k > 0) ? (s - 1.f) / (float)k : 0.f;
  }
  __syncthreads();
  float tau = taub[wv];
  for (int j = lane; j < d; j += 32) {
    float v = zb[wv][j] - tau;
    pe[csr[st + j]] = (v > 0.f) ? v : 0.f;
  }
}

// ----------------------------------------------------------------- GCN conv
__global__ void k_deg(const int* __restrict__ col, const float* __restrict__ pe,
                      float* __restrict__ deg) {
  int e = blockIdx.x * blockDim.x + threadIdx.x;
  if (e >= EE) return;
  float p = pe[e];
  if (p != 0.f) atomicAdd(&deg[col[e]], p);
}
__global__ void k_dinv(const float* __restrict__ deg, float* __restrict__ dinv) {
  int n = blockIdx.x * blockDim.x + threadIdx.x;
  if (n < NN) dinv[n] = rsqrtf(deg[n] + 1.f);  // +1 self-loop; always > 0
}
// out[col] += dinv[row]*p*dinv[col] * h[row];  one wave per edge, skip pruned.
__global__ void k_edge_agg(const int* __restrict__ row, const int* __restrict__ col,
                           const float* __restrict__ pe, const float* __restrict__ dinv,
                           const float* __restrict__ h, float* __restrict__ out) {
  int e = blockIdx.x * (blockDim.x >> 5) + (threadIdx.x >> 5);
  int lane = threadIdx.x & 31;
  if (e >= EE) return;
  float p = pe[e];
  if (p == 0.f) return;  // sparsemax pruned -> no HBM traffic
  int r = row[e], c = col[e];
  float nrm = dinv[r] * p * dinv[c];
  const float* hr = h + (size_t)r * HH;
  float* oc = out + (size_t)c * HH;
  __builtin_prefetch(hr, 0, 3);
#pragma unroll
  for (int f = lane; f < HH; f += 32) atomicAdd(&oc[f], nrm * hr[f]);
}
// relu(agg + h*dinv^2 + b) -> agg (f32) and bf16 copy for next GEMM
__global__ void k_conv_fin(float* __restrict__ agg, const float* __restrict__ h,
                           const float* __restrict__ dinv, const float* __restrict__ bias,
                           u16* __restrict__ outb) {
  long long i = blockIdx.x * (long long)blockDim.x + threadIdx.x;
  if (i >= (long long)NN * HH) return;
  int n = (int)(i / HH), f = (int)(i % HH);
  float di = dinv[n];
  float v = agg[i] + h[i] * di * di + bias[f];
  v = (v > 0.f) ? v : 0.f;
  agg[i] = v;
  outb[i] = f2bf(v);
}

// ----------------------------------------------------------------- pooling
__global__ void k_graph_count(const int* __restrict__ batch, int* __restrict__ cnt) {
  int n = blockIdx.x * blockDim.x + threadIdx.x;
  if (n < NN) atomicAdd(&cnt[batch[n]], 1);
}
__global__ void k_pool(const float* __restrict__ z, const int* __restrict__ batch,
                       float* __restrict__ feats, int blk) {
  long long i = blockIdx.x * (long long)blockDim.x + threadIdx.x;
  if (i >= (long long)NN * HH) return;
  int n = (int)(i / HH), f = (int)(i % HH);
  int g = batch[n];
  float v = z[i];
  atomicAdd(&feats[(size_t)g * FDIM + blk * 2 * HH + f], v);
  // z >= 0 after relu: uint bit order == float order, 0-init matches guard
  atomicMax((unsigned*)&feats[(size_t)g * FDIM + blk * 2 * HH + HH + f],
            __float_as_uint(v));
}
__global__ void k_feat_fin(float* __restrict__ feats, const int* __restrict__ cnt) {
  int i = blockIdx.x * blockDim.x + threadIdx.x;
  if (i >= GG * 3 * HH) return;
  int g = i / (3 * HH), rest = i % (3 * HH);
  int blk = rest / HH, f = rest % HH;
  int c = cnt[g];
  feats[(size_t)g * FDIM + blk * 2 * HH + f] /= (float)(c > 0 ? c : 1);
}

// ---------------------------------------------------------------- MLP head
__global__ void k_bias_relu(float* __restrict__ y, const float* __restrict__ bias,
                            u16* __restrict__ yb, int M, int Ncols) {
  int i = blockIdx.x * blockDim.x + threadIdx.x;
  if (i >= M * Ncols) return;
  float v = y[i] + bias[i % Ncols];
  v = (v > 0.f) ? v : 0.f;
  y[i] = v;
  if (yb) yb[i] = f2bf(v);
}
__global__ void k_fc3(const float* __restrict__ h, const float* __restrict__ w,
                      const float* __restrict__ b, float* __restrict__ out) {
  int i = blockIdx.x * blockDim.x + threadIdx.x;
  if (i >= GG * NCLS) return;
  int g = i / NCLS, c = i % NCLS;
  float s = b[c];
  const float* hr = h + (size_t)g * HH;
  for (int k = 0; k < HH; ++k) s += hr[k] * w[k * NCLS + c];
  out[i] = s;
}
__global__ void k_log_softmax(float* __restrict__ out) {
  int g = blockIdx.x * blockDim.x + threadIdx.x;
  if (g >= GG) return;
  float* r = out + (size_t)g * NCLS;
  float m = r[0];
  for (int j = 1; j < NCLS; ++j) m = fmaxf(m, r[j]);
  float s = 0.f;
  for (int j = 0; j < NCLS; ++j) s += __expf(r[j] - m);
  float ls = __logf(s);
  for (int j = 0; j < NCLS; ++j) r[j] = r[j] - m - ls;
}

// ------------------------------------------------------------------- host
static inline int cdiv(long long a, int b) { return (int)((a + b - 1) / b); }

extern "C" void kernel_launch(void* const* d_in, const int* in_sizes, int n_in,
                              void* d_out, int out_size, void* d_ws, size_t ws_size,
                              hipStream_t stream) {
  (void)in_sizes; (void)n_in; (void)out_size; (void)ws_size;
  const float* x     = (const float*)d_in[0];
  const int*   ei    = (const int*)d_in[1];      // [2,E]
  const int*   batch = (const int*)d_in[3];
  const float* att   = (const float*)d_in[5];    // [3, 2C]
  const float* W1    = (const float*)d_in[6];    // [3, C, H]
  const float* b1    = (const float*)d_in[7];    // [3, H]
  const float* W2    = (const float*)d_in[8];
  const float* b2    = (const float*)d_in[9];
  const float* fc1w  = (const float*)d_in[10];   // [6H, 2H]
  const float* fc1b  = (const float*)d_in[11];
  const float* fc2w  = (const float*)d_in[12];   // [2H, H]
  const float* fc2b  = (const float*)d_in[13];
  const float* fc3w  = (const float*)d_in[14];   // [H, NC]
  const float* fc3b  = (const float*)d_in[15];
  float* out = (float*)d_out;
  const int* row = ei;
  const int* col = ei + EE;

  // -------- workspace carve (256B aligned slabs)
  char* w = (char*)d_ws;
  auto carve = [&](size_t bytes) -> void* {
    void* p = (void*)w;
    w += (bytes + 255) & ~(size_t)255;
    return p;
  };
  u16*   xb     = (u16*)carve((size_t)NN * CC * 2);
  u16*   Wt     = (u16*)carve((size_t)CC * HH * 2);
  float* s1     = (float*)carve((size_t)NN * 4);
  float* s2     = (float*)carve((size_t)NN * 4);
  float* wsc    = (float*)carve((size_t)EE * 4);
  float* pe     = (float*)carve((size_t)EE * 4);
  int*   rc     = (int*)carve((size_t)NN * 4);
  int*   rstart = (int*)carve((size_t)NN * 4);
  int*   rcur   = (int*)carve((size_t)NN * 4);
  int*   csr    = (int*)carve((size_t)EE * 4);
  float* degw   = (float*)carve((size_t)NN * 4);
  float* dinvv  = (float*)carve((size_t)NN * 4);
  float* hmm    = (float*)carve((size_t)NN * HH * 4);
  float* agg    = (float*)carve((size_t)NN * HH * 4);
  u16*   actb   = (u16*)carve((size_t)NN * HH * 2);
  float* feats  = (float*)carve((size_t)GG * FDIM * 4);
  int*   cnt    = (int*)carve((size_t)GG * 4);
  u16*   featsb = (u16*)carve((size_t)GG * FDIM * 2);
  float* fc1o   = (float*)carve((size_t)GG * 2 * HH * 4);
  u16*   fc1ob  = (u16*)carve((size_t)GG * 2 * HH * 2);
  u16*   fcwT   = (u16*)carve((size_t)FDIM * 2 * HH * 2);  // big enough for fc1 & fc2
  float* fc2o   = (float*)carve((size_t)GG * HH * 4);

  const long long NH = (long long)NN * HH;

  // -------- one-time (per call) prep
  k_f32_to_bf16<<<4096, 256, 0, stream>>>(x, xb, (long long)NN * CC);
  k_zero_i<<<cdiv(NN, 256), 256, 0, stream>>>(rc, NN);
  k_row_count<<<cdiv(EE, 256), 256, 0, stream>>>(row, rc);
  k_scan_excl<<<1, 1024, 0, stream>>>(rc, rstart, NN);
  k_zero_i<<<cdiv(NN, 256), 256, 0, stream>>>(rcur, NN);
  k_csr_fill<<<cdiv(EE, 256), 256, 0, stream>>>(row, rstart, rcur, csr);
  k_zero_i<<<cdiv(GG, 256), 256, 0, stream>>>(cnt, GG);
  k_graph_count<<<cdiv(NN, 256), 256, 0, stream>>>(batch, cnt);
  k_zero_f<<<cdiv((long long)GG * FDIM, 256), 256, 0, stream>>>(feats,
                                                               (long long)GG * FDIM);

  // -------- three blocks
  for (int blk = 0; blk < 3; ++blk) {
    const float* attb = att + (size_t)blk * 2 * CC;
    // edge attention scores + leaky relu
    k_scores<<<cdiv(NN, 8), 256, 0, stream>>>(x, attb, s1, s2);
    k_edge_w<<<cdiv(EE, 256), 256, 0, stream>>>(row, col, s1, s2, wsc);
    // exact sparsemax per row-segment
    k_zero_f<<<4096, 256, 0, stream>>>(pe, EE);
    k_sparsemax<<<cdiv(NN, 8), 256, 0, stream>>>(wsc, rstart, rc, csr, pe);
    // edge-weight degree norm
    k_zero_f<<<cdiv(NN, 256), 256, 0, stream>>>(degw, NN);
    k_deg<<<cdiv(EE, 256), 256, 0, stream>>>(col, pe, degw);
    k_dinv<<<cdiv(NN, 256), 256, 0, stream>>>(degw, dinvv);

    // ---- conv1: h = x @ W1[blk]  (WMMA bf16), then sparse aggregate
    k_transpose_bf16<<<64, 256, 0, stream>>>(W1 + (size_t)blk * CC * HH, Wt, CC, HH);
    {
      dim3 g(NN / 16, HH / 16);
      k_wmma_gemm<<<g, 32, 0, stream>>>(xb, Wt, hmm, NN, HH, CC);
    }
    k_zero_f<<<8192, 256, 0, stream>>>(agg, NH);
    k_edge_agg<<<cdiv(EE, 8), 256, 0, stream>>>(row, col, pe, dinvv, hmm, agg);
    k_conv_fin<<<cdiv(NH, 256), 256, 0, stream>>>(agg, hmm, dinvv,
                                                  b1 + (size_t)blk * HH, actb);

    // ---- conv2: h = act @ W2[blk]
    k_transpose_bf16<<<64, 256, 0, stream>>>(W2 + (size_t)blk * HH * HH, Wt, HH, HH);
    {
      dim3 g(NN / 16, HH / 16);
      k_wmma_gemm<<<g, 32, 0, stream>>>(actb, Wt, hmm, NN, HH, HH);
    }
    k_zero_f<<<8192, 256, 0, stream>>>(agg, NH);
    k_edge_agg<<<cdiv(EE, 8), 256, 0, stream>>>(row, col, pe, dinvv, hmm, agg);
    k_conv_fin<<<cdiv(NH, 256), 256, 0, stream>>>(agg, hmm, dinvv,
                                                  b2 + (size_t)blk * HH, actb);

    // ---- global mean/max pooling into feats[:, blk*256 : blk*256+256]
    k_pool<<<cdiv(NH, 256), 256, 0, stream>>>(agg, batch, feats, blk);
  }

  // -------- head: fc1 (WMMA) -> fc2 (WMMA) -> fc3 -> log_softmax
  k_feat_fin<<<cdiv(GG * 3 * HH, 256), 256, 0, stream>>>(feats, cnt);
  k_f32_to_bf16<<<cdiv((long long)GG * FDIM, 256), 256, 0, stream>>>(
      feats, featsb, (long long)GG * FDIM);

  k_transpose_bf16<<<768, 256, 0, stream>>>(fc1w, fcwT, FDIM, 2 * HH);
  {
    dim3 g(GG / 16, (2 * HH) / 16);
    k_wmma_gemm<<<g, 32, 0, stream>>>(featsb, fcwT, fc1o, GG, 2 * HH, FDIM);
  }
  k_bias_relu<<<cdiv(GG * 2 * HH, 256), 256, 0, stream>>>(fc1o, fc1b, fc1ob, GG,
                                                          2 * HH);

  k_transpose_bf16<<<128, 256, 0, stream>>>(fc2w, fcwT, 2 * HH, HH);
  {
    dim3 g(GG / 16, HH / 16);
    k_wmma_gemm<<<g, 32, 0, stream>>>(fc1ob, fcwT, fc2o, GG, HH, 2 * HH);
  }
  k_bias_relu<<<cdiv(GG * HH, 256), 256, 0, stream>>>(fc2o, fc2b, (u16*)nullptr, GG,
                                                      HH);

  k_fc3<<<cdiv(GG * NCLS, 256), 256, 0, stream>>>(fc2o, fc3w, fc3b, out);
  k_log_softmax<<<cdiv(GG, 256), 256, 0, stream>>>(out);
}
